// StandardAttention_3032246911478
// MI455X (gfx1250) — matrix operand order repaired and checked by reading, not compiled, not run on hardware
//
#include <hip/hip_runtime.h>
#include <hip/hip_bf16.h>
#include <stdint.h>

// ---------------- types ----------------
typedef __attribute__((ext_vector_type(16))) __bf16 bf16x16;
typedef __attribute__((ext_vector_type(8)))  __bf16 bf16x8;
typedef __attribute__((ext_vector_type(8)))  float  f32x8;
typedef __attribute__((ext_vector_type(4)))  int    i32x4;

#define AS1 __attribute__((address_space(1)))
#define AS3 __attribute__((address_space(3)))

#if __has_builtin(__builtin_amdgcn_global_load_async_to_lds_b128) && \
    __has_builtin(__builtin_amdgcn_s_wait_asynccnt)
#define HAVE_ASYNC_LDS 1
#endif

__device__ __forceinline__ void async_copy16(const __bf16* gsrc, __bf16* ldst) {
#if defined(HAVE_ASYNC_LDS)
  __builtin_amdgcn_global_load_async_to_lds_b128(
      (AS1 i32x4*)(uintptr_t)gsrc, (AS3 i32x4*)(uintptr_t)ldst, 0, 0);
#else
  *(bf16x8*)ldst = *(const bf16x8*)gsrc;
#endif
}
__device__ __forceinline__ void async_wait0() {
#if defined(HAVE_ASYNC_LDS)
  __builtin_amdgcn_s_wait_asynccnt(0);
#endif
}

// ---------------- problem constants ----------------
#define BB   2
#define SS   2048
#define DM   2048
#define NH   16
#define NKV  4
#define HD   128
#define MROWS (BB*SS)          // 4096
#define LOG_ROPE_BASE 13.122363377404328f   // ln(500000)

// ================= cast f32 -> bf16 =================
__global__ void cast_f32_bf16(const float* __restrict__ in, __bf16* __restrict__ out, int n) {
  int i = blockIdx.x * blockDim.x + threadIdx.x;
  if (i < n) out[i] = (__bf16)in[i];
}

// ================= RoPE (+cast) =================
// layout: in/out [rows][H][128]; one thread handles pair (j, j+64)
__global__ void rope_cast(const float* __restrict__ in, __bf16* __restrict__ out,
                          int rows, int H, int applyRope) {
  int idx = blockIdx.x * blockDim.x + threadIdx.x;
  int total = rows * H * 64;
  if (idx >= total) return;
  int j   = idx & 63;
  int h   = (idx >> 6) % H;
  int row = idx / (64 * H);
  int s   = row & (SS - 1);
  const float* p = in  + ((size_t)row * H + h) * HD;
  __bf16*      q = out + ((size_t)row * H + h) * HD;
  float x1 = p[j], x2 = p[j + 64];
  if (applyRope) {
    float invf = __expf(-(float)j * (LOG_ROPE_BASE / 64.0f));
    float ang  = (float)s * invf;
    float c = cosf(ang), sn = sinf(ang);
    q[j]      = (__bf16)(x1 * c - x2 * sn);
    q[j + 64] = (__bf16)(x2 * c + x1 * sn);
  } else {
    q[j]      = (__bf16)x1;
    q[j + 64] = (__bf16)x2;
  }
}

// ================= bf16 WMMA GEMM: C(f32) = A(bf16,MxK) * B(bf16,KxN) =================
// 256 threads (8 waves, 4x2), block tile 128x128, BK=32.
// wave (wm,wn): rows wm*32..+32, cols wn*64..+64 -> 2 A-frags, 4 B-frags, 8 WMMAs/K-step.
#define GTM 128
#define GTN 128
#define GBK 32
__global__ __launch_bounds__(256) void gemm_bf16_wmma(
    const __bf16* __restrict__ A, const __bf16* __restrict__ B,
    float* __restrict__ C, int M, int N, int K) {
  __shared__ alignas(32) __bf16 lA [GTM * GBK];   // [row][k]   8KB
  __shared__ alignas(32) __bf16 lBt[GTN * GBK];   // [col][k]   8KB (transposed)
  const int tid  = threadIdx.x;
  const int wave = tid >> 5;
  const int lane = tid & 31;
  const int g    = lane >> 4;
  const int l16  = lane & 15;
  const int wm   = wave >> 1;     // 0..3
  const int wn   = wave & 1;      // 0..1
  const int mt   = blockIdx.y * GTM;
  const int nt   = blockIdx.x * GTN;

  f32x8 acc[2][4] = {};
  for (int kb = 0; kb < K; kb += GBK) {
    __syncthreads();
    { // A tile 128x32: 16 elems/thread, natural layout -> async global->LDS
      int row = tid >> 1;
      int k0  = (tid & 1) * 16;
      const __bf16* src = A + (size_t)(mt + row) * K + kb + k0;
      async_copy16(src,     &lA[row * GBK + k0]);
      async_copy16(src + 8, &lA[row * GBK + k0 + 8]);
    }
    { // B tile 32x128: 16 elems/thread, stored transposed (VALU path)
      int kr = tid >> 3;            // 0..31
      int c0 = (tid & 7) * 16;
      const __bf16* src = B + (size_t)(kb + kr) * N + nt + c0;
      bf16x8 b0 = *(const bf16x8*)(src);
      bf16x8 b1 = *(const bf16x8*)(src + 8);
#pragma unroll
      for (int i = 0; i < 8; ++i) lBt[(c0 + i) * GBK + kr]     = b0[i];
#pragma unroll
      for (int i = 0; i < 8; ++i) lBt[(c0 + 8 + i) * GBK + kr] = b1[i];
    }
    async_wait0();
    __syncthreads();

    // Load all fragments first, then issue 8 WMMAs (batched waits, better ILP).
    bf16x16 af[2];
#pragma unroll
    for (int i = 0; i < 2; ++i) {
      int row = wm * 32 + i * 16 + l16;
      bf16x8 lo = *(const bf16x8*)&lA[row * GBK + g * 8];
      bf16x8 hi = *(const bf16x8*)&lA[row * GBK + 16 + g * 8];
#pragma unroll
      for (int e = 0; e < 8; ++e) { af[i][e] = lo[e]; af[i][8 + e] = hi[e]; }
    }
    bf16x16 bfr[4];
#pragma unroll
    for (int t = 0; t < 4; ++t)
    {
      bf16x8 lo = *(const bf16x8*)&lBt[(wn * 64 + t * 16 + l16) * GBK + g * 8];
      bf16x8 hi = *(const bf16x8*)&lBt[(wn * 64 + t * 16 + l16) * GBK + 16 + g * 8];
#pragma unroll
      for (int e = 0; e < 8; ++e) { bfr[t][e] = lo[e]; bfr[t][8 + e] = hi[e]; }
    }

#pragma unroll
    for (int i = 0; i < 2; ++i)
#pragma unroll
      for (int t = 0; t < 4; ++t)
        acc[i][t] = __builtin_amdgcn_wmma_f32_16x16x32_bf16(
            false, af[i], false, bfr[t], (short)0, acc[i][t], false, false);
  }
  // C/D layout: VGPR r -> row r + 8*g, col = l16
#pragma unroll
  for (int i = 0; i < 2; ++i)
#pragma unroll
    for (int t = 0; t < 4; ++t)
#pragma unroll
      for (int r = 0; r < 8; ++r) {
        int row = mt + wm * 32 + i * 16 + r + 8 * g;
        int col = nt + wn * 64 + t * 16 + l16;
        C[(size_t)row * N + col] = acc[i][t][r];
      }
}

// ================= flash-style causal GQA attention =================
// Q: [B][S][NH][HD] bf16 (roped)   K,V: [B][S][NKV][HD] bf16
// out: [B][S][NH][HD] bf16.  Block = 128 thr (4 waves) handles 64 q-rows of one (b,h).
__global__ __launch_bounds__(128) void attn_fwd(
    const __bf16* __restrict__ Q, const __bf16* __restrict__ Kk,
    const __bf16* __restrict__ V, __bf16* __restrict__ O) {
  __shared__ alignas(32) __bf16 lK [32 * HD];      // [key][d]
  __shared__ alignas(32) __bf16 lVt[HD * 32];      // [d][key] (transposed)
  __shared__ alignas(32) __bf16 lP [4][16 * 32];   // per-wave P tile
  const int tid  = threadIdx.x;
  const int wave = tid >> 5;
  const int lane = tid & 31;
  const int g    = lane >> 4;
  const int l16  = lane & 15;
  const int qb = blockIdx.x, h = blockIdx.y, b = blockIdx.z;
  const int kvh = h >> 2;
  const int qrow0 = qb * 64 + wave * 16;
  const float scale = 0.08838834764831845f;  // 1/sqrt(128)

  // Q fragments: 4 chunks of K=32 covering d=0..127
  bf16x16 qf[4];
  {
    const __bf16* qp = Q + (((size_t)(b * SS + qrow0 + l16)) * NH + h) * HD;
#pragma unroll
    for (int c = 0; c < 4; ++c) {
      bf16x8 lo = *(const bf16x8*)(qp + c * 32 + g * 8);
      bf16x8 hi = *(const bf16x8*)(qp + c * 32 + 16 + g * 8);
#pragma unroll
      for (int i = 0; i < 8; ++i) { qf[c][i] = lo[i]; qf[c][8 + i] = hi[i]; }
    }
  }

  f32x8 oacc[8] = {};
  float mrow[8], lrow[8];
#pragma unroll
  for (int r = 0; r < 8; ++r) { mrow[r] = -INFINITY; lrow[r] = 0.f; }

  const int nkb = 2 * qb + 2;  // uniform over block: keys 0 .. qb*64+63
  for (int kb = 0; kb < nkb; ++kb) {
    __syncthreads();
    {  // stage K (async, natural) and V (transposed), 32 keys x 128 d
      int row = tid >> 2;
      int c0  = (tid & 3) * 32;
      const __bf16* kp = Kk + (((size_t)(b * SS + kb * 32 + row)) * NKV + kvh) * HD + c0;
      const __bf16* vp = V  + (((size_t)(b * SS + kb * 32 + row)) * NKV + kvh) * HD + c0;
#pragma unroll
      for (int i = 0; i < 4; ++i)
        async_copy16(kp + i * 8, &lK[row * HD + c0 + i * 8]);
      bf16x8 vv[4];
#pragma unroll
      for (int i = 0; i < 4; ++i) vv[i] = *(const bf16x8*)(vp + i * 8);
#pragma unroll
      for (int i = 0; i < 4; ++i)
#pragma unroll
        for (int j = 0; j < 8; ++j) lVt[(c0 + i * 8 + j) * 32 + row] = vv[i][j];
    }
    async_wait0();
    __syncthreads();

    // scores: S0 (keys kb*32..+15), S1 (+16..+31); contraction over d in 4 chunks
    f32x8 s0 = {}, s1 = {};
#pragma unroll
    for (int c = 0; c < 4; ++c) {
      bf16x16 k0, k1;
      {
        bf16x8 a0 = *(const bf16x8*)&lK[l16 * HD + c * 32 + g * 8];
        bf16x8 a1 = *(const bf16x8*)&lK[l16 * HD + c * 32 + 16 + g * 8];
        bf16x8 b0 = *(const bf16x8*)&lK[(16 + l16) * HD + c * 32 + g * 8];
        bf16x8 b1 = *(const bf16x8*)&lK[(16 + l16) * HD + c * 32 + 16 + g * 8];
#pragma unroll
        for (int i = 0; i < 8; ++i) { k0[i] = a0[i]; k0[8 + i] = a1[i]; k1[i] = b0[i]; k1[8 + i] = b1[i]; }
      }
      s0 = __builtin_amdgcn_wmma_f32_16x16x32_bf16(false, qf[c], false, k0, (short)0, s0, false, false);
      s1 = __builtin_amdgcn_wmma_f32_16x16x32_bf16(false, qf[c], false, k1, (short)0, s1, false, false);
    }

    // causal mask + online softmax (row r of this wave = qrow0 + r + 8*g)
    const int qg = qrow0 + 8 * g;
    const int kc0 = kb * 32 + l16;
    const int kc1 = kc0 + 16;
    float alpha[8];
#pragma unroll
    for (int r = 0; r < 8; ++r) {
      float v0 = s0[r] * scale, v1 = s1[r] * scale;
      if (kc0 > qg + r) v0 = -INFINITY;
      if (kc1 > qg + r) v1 = -INFINITY;
      float mx = fmaxf(v0, v1);
#pragma unroll
      for (int msk = 1; msk < 16; msk <<= 1) mx = fmaxf(mx, __shfl_xor(mx, msk, 32));
      float mnew = fmaxf(mrow[r], mx);
      float a  = __expf(mrow[r] - mnew);
      float p0 = __expf(v0 - mnew);
      float p1 = __expf(v1 - mnew);
      float rs = p0 + p1;
#pragma unroll
      for (int msk = 1; msk < 16; msk <<= 1) rs += __shfl_xor(rs, msk, 32);
      lrow[r] = lrow[r] * a + rs;
      mrow[r] = mnew;
      alpha[r] = a;
      lP[wave][(r + 8 * g) * 32 + l16]      = (__bf16)p0;
      lP[wave][(r + 8 * g) * 32 + 16 + l16] = (__bf16)p1;
    }
#pragma unroll
    for (int t = 0; t < 8; ++t)
#pragma unroll
      for (int r = 0; r < 8; ++r) oacc[t][r] *= alpha[r];
    __syncthreads();

    // P fragment (16x32) from LDS
    bf16x16 pf;
    {
      const __bf16* pp = &lP[wave][l16 * 32];
      bf16x8 lo = *(const bf16x8*)(pp + g * 8);
      bf16x8 hi = *(const bf16x8*)(pp + 16 + g * 8);
#pragma unroll
      for (int i = 0; i < 8; ++i) { pf[i] = lo[i]; pf[8 + i] = hi[i]; }
    }
    // O += P(16x32) * V(32x128)
#pragma unroll
    for (int t = 0; t < 8; ++t) {
      bf16x16 vfr;
      {
        bf16x8 lo = *(const bf16x8*)&lVt[(t * 16 + l16) * 32 + g * 8];
        bf16x8 hi = *(const bf16x8*)&lVt[(t * 16 + l16) * 32 + 16 + g * 8];
#pragma unroll
        for (int i = 0; i < 8; ++i) { vfr[i] = lo[i]; vfr[8 + i] = hi[i]; }
      }
      oacc[t] = __builtin_amdgcn_wmma_f32_16x16x32_bf16(
          false, pf, false, vfr, (short)0, oacc[t], false, false);
    }
  }

  // epilogue: O /= l, store bf16 at [b][q][h][d]
#pragma unroll
  for (int r = 0; r < 8; ++r) {
    float inv = 1.0f / lrow[r];
    int qrow = qrow0 + r + 8 * g;
    __bf16* op = O + (((size_t)(b * SS + qrow)) * NH + h) * HD;
#pragma unroll
    for (int t = 0; t < 8; ++t) op[t * 16 + l16] = (__bf16)(oacc[t][r] * inv);
  }
}

// ================= host launch =================
extern "C" void kernel_launch(void* const* d_in, const int* in_sizes, int n_in,
                              void* d_out, int out_size, void* d_ws, size_t ws_size,
                              hipStream_t stream) {
  const float* x  = (const float*)d_in[0];
  const float* Wq = (const float*)d_in[1];
  const float* Wk = (const float*)d_in[2];
  const float* Wv = (const float*)d_in[3];
  const float* Wo = (const float*)d_in[4];
  float* out = (float*)d_out;

  char* ws = (char*)d_ws;
  size_t off = 0;
  auto alloc = [&](size_t bytes) { void* p = ws + off; off = (off + bytes + 255) & ~(size_t)255; return p; };

  __bf16* Xb  = (__bf16*)alloc((size_t)MROWS * DM * 2);
  __bf16* Wqb = (__bf16*)alloc((size_t)DM * (NH * HD) * 2);
  __bf16* Wkb = (__bf16*)alloc((size_t)DM * (NKV * HD) * 2);
  __bf16* Wvb = (__bf16*)alloc((size_t)DM * (NKV * HD) * 2);
  __bf16* Wob = (__bf16*)alloc((size_t)(NH * HD) * DM * 2);
  float*  Qf  = (float*)alloc((size_t)MROWS * (NH * HD) * 4);
  float*  Kf  = (float*)alloc((size_t)MROWS * (NKV * HD) * 4);
  float*  Vf  = (float*)alloc((size_t)MROWS * (NKV * HD) * 4);
  __bf16* Qb  = (__bf16*)alloc((size_t)MROWS * (NH * HD) * 2);
  __bf16* Kb  = (__bf16*)alloc((size_t)MROWS * (NKV * HD) * 2);
  __bf16* Vb  = (__bf16*)alloc((size_t)MROWS * (NKV * HD) * 2);
  __bf16* Ab  = (__bf16*)alloc((size_t)MROWS * (NH * HD) * 2);

  auto cast = [&](const float* src, __bf16* dst, int n) {
    cast_f32_bf16<<<(n + 255) / 256, 256, 0, stream>>>(src, dst, n);
  };
  cast(x,  Xb,  MROWS * DM);
  cast(Wq, Wqb, DM * NH * HD);
  cast(Wk, Wkb, DM * NKV * HD);
  cast(Wv, Wvb, DM * NKV * HD);
  cast(Wo, Wob, NH * HD * DM);

  // QKV projections
  gemm_bf16_wmma<<<dim3((NH * HD) / GTN, MROWS / GTM), 256, 0, stream>>>(
      Xb, Wqb, Qf, MROWS, NH * HD, DM);
  gemm_bf16_wmma<<<dim3((NKV * HD) / GTN, MROWS / GTM), 256, 0, stream>>>(
      Xb, Wkb, Kf, MROWS, NKV * HD, DM);
  gemm_bf16_wmma<<<dim3((NKV * HD) / GTN, MROWS / GTM), 256, 0, stream>>>(
      Xb, Wvb, Vf, MROWS, NKV * HD, DM);

  // RoPE + casts to bf16
  {
    int nq = MROWS * NH * 64;
    rope_cast<<<(nq + 255) / 256, 256, 0, stream>>>(Qf, Qb, MROWS, NH, 1);
    int nk = MROWS * NKV * 64;
    rope_cast<<<(nk + 255) / 256, 256, 0, stream>>>(Kf, Kb, MROWS, NKV, 1);
    rope_cast<<<(nk + 255) / 256, 256, 0, stream>>>(Vf, Vb, MROWS, NKV, 0);
  }

  // causal flash attention
  attn_fwd<<<dim3(SS / 64, NH, BB), 128, 0, stream>>>(Qb, Kb, Vb, Ab);

  // output projection -> f32 out
  gemm_bf16_wmma<<<dim3(DM / GTN, MROWS / GTM), 256, 0, stream>>>(
      Ab, Wob, out, MROWS, DM, NH * HD);
}
